// BasicTransformerBlock_86818468921693
// MI455X (gfx1250) — compile-verified, matrix operand order checked by reading, not run on hardware
//
#include <hip/hip_runtime.h>
#include <hip/hip_bf16.h>
#include <math.h>
#include <stdint.h>

typedef __bf16 bf16_t;
typedef __attribute__((ext_vector_type(16))) __bf16 v16bf;
typedef __attribute__((ext_vector_type(8)))  __bf16 v8bf;
typedef __attribute__((ext_vector_type(8)))  float  v8f;

#define D_MODEL 1280
#define N_TOK   256
#define BF_TOT  32
#define F_VID   16
#define HEADS   20
#define DH      64
#define M_TOT   (BF_TOT * N_TOK)   // 8192
#define DC_ENC  768
#define NK_CROSS 77
#define ME_TOT  (BF_TOT * NK_CROSS) // 2464
#define FF_HALF 5120
#define QK_SCALE 0.125f            // 64^-0.5

// ---------------------------------------------------------------------------
// helpers
// ---------------------------------------------------------------------------
__device__ __forceinline__ v8f wmma_bf16(v16bf a, v16bf b, v8f c) {
    return __builtin_amdgcn_wmma_f32_16x16x32_bf16(false, a, false, b, (short)0, c, false, false);
}

// Build a 16x32 (or 32x16) bf16 fragment for one lane from a row pointer whose
// K elements are contiguous.  Per ISA layout: VGPR0..3 hold K = kb..kb+7,
// VGPR4..7 hold K = kb+16..kb+23 (kb = 0 for lanes 0-15, 8 for lanes 16-31).
// The caller passes p = &row[kb]; we read p[0..7] and p[16..23].
__device__ __forceinline__ v16bf frag16(const bf16_t* p) {
    v8bf lo = *(const v8bf*)p;
    v8bf hi = *(const v8bf*)(p + 16);
    v16bf f;
#pragma unroll
    for (int e = 0; e < 8; ++e) { f[e] = lo[e]; f[8 + e] = hi[e]; }
    return f;
}

__device__ __forceinline__ v16bf frag16_scaled(const bf16_t* p, float s) {
    v8bf lo = *(const v8bf*)p;
    v8bf hi = *(const v8bf*)(p + 16);
    v16bf f;
#pragma unroll
    for (int e = 0; e < 8; ++e) {
        f[e]     = (bf16_t)((float)lo[e] * s);
        f[8 + e] = (bf16_t)((float)hi[e] * s);
    }
    return f;
}

__device__ __forceinline__ float redmax16(float v) {
#pragma unroll
    for (int d = 8; d; d >>= 1) v = fmaxf(v, __shfl_xor(v, d, 16));
    return v;
}
__device__ __forceinline__ float redsum16(float v) {
#pragma unroll
    for (int d = 8; d; d >>= 1) v += __shfl_xor(v, d, 16);
    return v;
}

// ---------------------------------------------------------------------------
// conversion kernels
// ---------------------------------------------------------------------------
__global__ void cvt_kernel(const float* __restrict__ in, bf16_t* __restrict__ out, int n) {
    int i = blockIdx.x * 256 + threadIdx.x;
    if (i < n) out[i] = (bf16_t)in[i];
}

// fp32 W[K][N] row-major -> bf16 W^T[N][K] row-major
__global__ void cvtT_kernel(const float* __restrict__ in, bf16_t* __restrict__ out, int K, int N) {
    int i = blockIdx.x * 256 + threadIdx.x;
    if (i < K * N) {
        int n = i / K, k = i % K;
        out[i] = (bf16_t)in[(size_t)k * N + n];
    }
}

// ---------------------------------------------------------------------------
// LayerNorm: one row (length D) per 256-thread block, bf16 output
// ---------------------------------------------------------------------------
__global__ __launch_bounds__(256) void ln_kernel(const float* __restrict__ x,
                                                 const float* __restrict__ w,
                                                 const float* __restrict__ b,
                                                 bf16_t* __restrict__ out, int D) {
    int row = blockIdx.x, t = threadIdx.x;
    const float* xr = x + (size_t)row * D;
    float s = 0.f, s2 = 0.f;
    for (int i = t; i < D; i += 256) { float v = xr[i]; s += v; s2 += v * v; }
#pragma unroll
    for (int off = 16; off; off >>= 1) { s += __shfl_xor(s, off, 32); s2 += __shfl_xor(s2, off, 32); }
    __shared__ float rs[8], rs2[8];
    if ((t & 31) == 0) { rs[t >> 5] = s; rs2[t >> 5] = s2; }
    __syncthreads();
    float ts = 0.f, ts2 = 0.f;
#pragma unroll
    for (int i = 0; i < 8; ++i) { ts += rs[i]; ts2 += rs2[i]; }
    float mean = ts / (float)D;
    float var  = ts2 / (float)D - mean * mean;
    float inv  = rsqrtf(var + 1e-5f);
    for (int i = t; i < D; i += 256)
        out[(size_t)row * D + i] = (bf16_t)((xr[i] - mean) * inv * w[i] + b[i]);
}

// ---------------------------------------------------------------------------
// GEMM: C[M,N] = A[M,K](bf16) @ B (given as B^T[N,K] bf16) + bias + residual
// 128x128 tile, 4 waves of 64x64, v_wmma_f32_16x16x32_bf16.
// Register-staged double-buffered LDS: one barrier per K-step, global loads
// for step t+1 issued before the WMMA block of step t, prefetch for t+2.
// ---------------------------------------------------------------------------
__global__ __launch_bounds__(128) void gemm_kernel(const bf16_t* __restrict__ A,
                                                   const bf16_t* __restrict__ BT,
                                                   const float* __restrict__ bias,
                                                   const float* __restrict__ resid,
                                                   float* __restrict__ Cf,
                                                   bf16_t* __restrict__ Cb,
                                                   int M, int N, int K) {
    __shared__ bf16_t As[2][128][40];
    __shared__ bf16_t Bs[2][128][40];
    const int tid = threadIdx.x, lane = tid & 31, wv = tid >> 5;
    const int bm = blockIdx.y * 128, bn = blockIdx.x * 128;
    const int wm = (wv >> 1) * 64, wn = (wv & 1) * 64;
    const int fl = lane & 15, kb = (lane >> 4) * 8, rofs = 8 * (lane >> 4);
    const int r = tid;
    const int gmA = bm + r;
    const bool fullM = (bm + 128 <= M);
    const bool aValid = fullM || (gmA < M);
    const bf16_t* ap = A + (size_t)gmA * K;
    const bf16_t* bp = BT + (size_t)(bn + r) * K;

    v8bf ra[4], rb[4];
    auto loadTile = [&](int k0) {
        if (aValid) {
#pragma unroll
            for (int c = 0; c < 4; ++c) ra[c] = *(const v8bf*)(ap + k0 + c * 8);
        } else {
            v8bf z = {};
#pragma unroll
            for (int c = 0; c < 4; ++c) ra[c] = z;
        }
#pragma unroll
        for (int c = 0; c < 4; ++c) rb[c] = *(const v8bf*)(bp + k0 + c * 8);
    };
    auto storeTile = [&](int s) {
#pragma unroll
        for (int c = 0; c < 4; ++c) {
            *(v8bf*)&As[s][r][c * 8] = ra[c];
            *(v8bf*)&Bs[s][r][c * 8] = rb[c];
        }
    };

    v8f acc[4][4] = {};
    const int T = K >> 5;
    loadTile(0);
    storeTile(0);
    __syncthreads();
    for (int t = 0; t < T; ++t) {
        const int s = t & 1;
        if (t + 1 < T) {
            loadTile((t + 1) << 5);
            if (t + 2 < T) {
                if (aValid) __builtin_prefetch(ap + ((t + 2) << 5), 0, 1);
                __builtin_prefetch(bp + ((t + 2) << 5), 0, 1);
            }
        }
        v16bf af[4], bfb[4];
#pragma unroll
        for (int i = 0; i < 4; ++i) af[i] = frag16(&As[s][wm + i * 16 + fl][kb]);
#pragma unroll
        for (int j = 0; j < 4; ++j) bfb[j] = frag16(&Bs[s][wn + j * 16 + fl][kb]);
#pragma unroll
        for (int i = 0; i < 4; ++i)
#pragma unroll
            for (int j = 0; j < 4; ++j)
                acc[i][j] = wmma_bf16(af[i], bfb[j], acc[i][j]);
        if (t + 1 < T) storeTile(s ^ 1);
        __syncthreads();
    }
#pragma unroll
    for (int i = 0; i < 4; ++i)
#pragma unroll
        for (int j = 0; j < 4; ++j) {
            int gn = bn + wn + j * 16 + fl;
            float bv = bias ? bias[gn] : 0.f;
#pragma unroll
            for (int r8 = 0; r8 < 8; ++r8) {
                int gm = bm + wm + i * 16 + rofs + r8;
                if (fullM || gm < M) {
                    size_t idx = (size_t)gm * N + gn;
                    float v = acc[i][j][r8] + bv;
                    if (resid) v += resid[idx];
                    if (Cf) Cf[idx] = v;
                    if (Cb) Cb[idx] = (bf16_t)v;
                }
            }
        }
}

// ---------------------------------------------------------------------------
// GEGLU GEMM: hh = A@W[:, :N], gate = A@W[:, N:2N]; out = hh * gelu(gate)
// BT holds W^T[2N][K].  bias has length 2N.  Same pipelined structure.
// ---------------------------------------------------------------------------
__global__ __launch_bounds__(128) void geglu_kernel(const bf16_t* __restrict__ A,
                                                    const bf16_t* __restrict__ BT,
                                                    const float* __restrict__ bias,
                                                    bf16_t* __restrict__ Cb,
                                                    int M, int N, int K) {
    __shared__ bf16_t As[2][128][40];
    __shared__ bf16_t Bh[2][128][40];
    __shared__ bf16_t Bg[2][128][40];
    const int tid = threadIdx.x, lane = tid & 31, wv = tid >> 5;
    const int bm = blockIdx.y * 128, bn = blockIdx.x * 128;
    const int wm = (wv >> 1) * 64, wn = (wv & 1) * 64;
    const int fl = lane & 15, kb = (lane >> 4) * 8, rofs = 8 * (lane >> 4);
    const int r = tid;
    const bf16_t* ap  = A + (size_t)(bm + r) * K;
    const bf16_t* bph = BT + (size_t)(bn + r) * K;
    const bf16_t* bpg = BT + (size_t)(N + bn + r) * K;

    v8bf ra[4], rh[4], rg[4];
    auto loadTile = [&](int k0) {
#pragma unroll
        for (int c = 0; c < 4; ++c) {
            ra[c] = *(const v8bf*)(ap + k0 + c * 8);
            rh[c] = *(const v8bf*)(bph + k0 + c * 8);
            rg[c] = *(const v8bf*)(bpg + k0 + c * 8);
        }
    };
    auto storeTile = [&](int s) {
#pragma unroll
        for (int c = 0; c < 4; ++c) {
            *(v8bf*)&As[s][r][c * 8] = ra[c];
            *(v8bf*)&Bh[s][r][c * 8] = rh[c];
            *(v8bf*)&Bg[s][r][c * 8] = rg[c];
        }
    };

    v8f acch[4][4] = {}, accg[4][4] = {};
    const int T = K >> 5;
    loadTile(0);
    storeTile(0);
    __syncthreads();
    for (int t = 0; t < T; ++t) {
        const int s = t & 1;
        if (t + 1 < T) {
            loadTile((t + 1) << 5);
            if (t + 2 < T) {
                __builtin_prefetch(ap + ((t + 2) << 5), 0, 1);
                __builtin_prefetch(bph + ((t + 2) << 5), 0, 1);
                __builtin_prefetch(bpg + ((t + 2) << 5), 0, 1);
            }
        }
        v16bf af[4], bh[4], bg[4];
#pragma unroll
        for (int i = 0; i < 4; ++i) af[i] = frag16(&As[s][wm + i * 16 + fl][kb]);
#pragma unroll
        for (int j = 0; j < 4; ++j) {
            bh[j] = frag16(&Bh[s][wn + j * 16 + fl][kb]);
            bg[j] = frag16(&Bg[s][wn + j * 16 + fl][kb]);
        }
#pragma unroll
        for (int i = 0; i < 4; ++i)
#pragma unroll
            for (int j = 0; j < 4; ++j) {
                acch[i][j] = wmma_bf16(af[i], bh[j], acch[i][j]);
                accg[i][j] = wmma_bf16(af[i], bg[j], accg[i][j]);
            }
        if (t + 1 < T) storeTile(s ^ 1);
        __syncthreads();
    }
#pragma unroll
    for (int i = 0; i < 4; ++i)
#pragma unroll
        for (int j = 0; j < 4; ++j) {
            int gn = bn + wn + j * 16 + fl;
            float bhv = bias[gn], bgv = bias[N + gn];
#pragma unroll
            for (int r8 = 0; r8 < 8; ++r8) {
                int gm = bm + wm + i * 16 + rofs + r8;
                float hv = acch[i][j][r8] + bhv;
                float gv = accg[i][j][r8] + bgv;
                float ge = 0.5f * gv * (1.f + erff(gv * 0.70710678118654752f));
                Cb[(size_t)gm * N + gn] = (bf16_t)(hv * ge);
            }
        }
}

// ---------------------------------------------------------------------------
// Flash attention.  Block = (bf, head), 4 waves, 64 queries per wave.
// mode 0: sparse-causal self-attn, nk=512, KV rows gathered (frame0 + prev frame)
// mode 1: cross-attn, nk=77, KV rows = bf*77 + j
// ---------------------------------------------------------------------------
__global__ __launch_bounds__(128) void attn_kernel(const bf16_t* __restrict__ q,
                                                   const bf16_t* __restrict__ kbuf,
                                                   const bf16_t* __restrict__ vbuf,
                                                   bf16_t* __restrict__ o,
                                                   int mode, int nk) {
    const int BFi = blockIdx.x;   // 0..31
    const int h   = blockIdx.y;   // 0..19
    const int tid = threadIdx.x, lane = tid & 31, wv = tid >> 5;
    const int fl = lane & 15, kb = (lane >> 4) * 8, rofs = 8 * (lane >> 4);
    __shared__ bf16_t Ks[64][72];      // [key][dh]
    __shared__ bf16_t Vs[64][72];      // [dh][key]  (transposed)
    __shared__ bf16_t Ps[4][16][72];   // per-wave P strip [query][key]

    // Q fragments (pre-scaled), kept in registers for the whole kernel
    v16bf qf[4][2];
#pragma unroll
    for (int mi = 0; mi < 4; ++mi)
#pragma unroll
        for (int kc = 0; kc < 2; ++kc) {
            const bf16_t* p = q + ((size_t)(BFi * N_TOK + wv * 64 + mi * 16 + fl)) * D_MODEL
                                + h * DH + kc * 32 + kb;
            qf[mi][kc] = frag16_scaled(p, QK_SCALE);
        }

    float mstat[4][8], lsum[4][8];
#pragma unroll
    for (int i = 0; i < 4; ++i)
#pragma unroll
        for (int r = 0; r < 8; ++r) { mstat[i][r] = -1e30f; lsum[i][r] = 0.f; }
    v8f acc[4][4] = {};

    const int nt = (nk + 63) / 64;
    for (int ti = 0; ti < nt; ++ti) {
        const int t0 = ti * 64;
        __syncthreads();
        // stage K (row-major) and V (transposed) tiles
        for (int c = tid; c < 512; c += 128) {
            int row = c >> 3, col0 = (c & 7) * 8;
            int j = t0 + row;
            v8bf kv = {}, vv = {};
            if (j < nk) {
                size_t grow;
                if (mode == 0) {
                    int b = BFi >> 4, fi = BFi & 15;
                    int fsrc = (j < N_TOK) ? 0 : ((fi == 0) ? 0 : fi - 1);
                    int n = (j < N_TOK) ? j : j - N_TOK;
                    grow = (size_t)((b * F_VID + fsrc) * N_TOK + n);
                } else {
                    grow = (size_t)(BFi * NK_CROSS + j);
                }
                kv = *(const v8bf*)(kbuf + grow * D_MODEL + h * DH + col0);
                vv = *(const v8bf*)(vbuf + grow * D_MODEL + h * DH + col0);
            }
            *(v8bf*)&Ks[row][col0] = kv;
#pragma unroll
            for (int e = 0; e < 8; ++e) Vs[col0 + e][row] = vv[e];
        }
        __syncthreads();
        // K fragments (B layout: lane = key) and V fragments (B layout: lane = dh)
        v16bf kf[4][2], vf[4][2];
#pragma unroll
        for (int nj = 0; nj < 4; ++nj)
#pragma unroll
            for (int kc = 0; kc < 2; ++kc) {
                kf[nj][kc] = frag16(&Ks[nj * 16 + fl][kc * 32 + kb]);
                vf[nj][kc] = frag16(&Vs[nj * 16 + fl][kc * 32 + kb]);
            }

        for (int mi = 0; mi < 4; ++mi) {
            // scores: 16 queries x 64 keys
            v8f S[4];
#pragma unroll
            for (int nj = 0; nj < 4; ++nj) {
                v8f s = {};
                s = wmma_bf16(qf[mi][0], kf[nj][0], s);
                s = wmma_bf16(qf[mi][1], kf[nj][1], s);
                S[nj] = s;
            }
            if (mode == 1) {
#pragma unroll
                for (int nj = 0; nj < 4; ++nj) {
                    int kidx = t0 + nj * 16 + fl;
                    if (kidx >= nk) {
#pragma unroll
                        for (int r = 0; r < 8; ++r) S[nj][r] = -1e30f;
                    }
                }
            }
            // online softmax update
#pragma unroll
            for (int r = 0; r < 8; ++r) {
                float tm = fmaxf(fmaxf(S[0][r], S[1][r]), fmaxf(S[2][r], S[3][r]));
                tm = redmax16(tm);
                float mo = mstat[mi][r];
                float mn = fmaxf(mo, tm);
                mstat[mi][r] = mn;
                float al = expf(mo - mn);
                float ts = 0.f;
#pragma unroll
                for (int nj = 0; nj < 4; ++nj) {
                    float pv = expf(S[nj][r] - mn);
                    S[nj][r] = pv;
                    ts += pv;
                }
                ts = redsum16(ts);
                lsum[mi][r] = lsum[mi][r] * al + ts;
#pragma unroll
                for (int nj = 0; nj < 4; ++nj) acc[mi][nj][r] *= al;
#pragma unroll
                for (int nj = 0; nj < 4; ++nj)
                    Ps[wv][rofs + r][nj * 16 + fl] = (bf16_t)S[nj][r];
            }
            __syncthreads();
            // O += P @ V
#pragma unroll
            for (int kc = 0; kc < 2; ++kc) {
                v16bf pf = frag16(&Ps[wv][fl][kc * 32 + kb]);
#pragma unroll
                for (int nj = 0; nj < 4; ++nj)
                    acc[mi][nj] = wmma_bf16(pf, vf[nj][kc], acc[mi][nj]);
            }
        }
    }
    // normalize and store
#pragma unroll
    for (int mi = 0; mi < 4; ++mi)
#pragma unroll
        for (int nj = 0; nj < 4; ++nj)
#pragma unroll
            for (int r = 0; r < 8; ++r) {
                size_t grow = (size_t)BFi * N_TOK + wv * 64 + mi * 16 + rofs + r;
                float ov = acc[mi][nj][r] / lsum[mi][r];
                o[grow * D_MODEL + (size_t)h * DH + nj * 16 + fl] = (bf16_t)ov;
            }
}

// ---------------------------------------------------------------------------
// Temporal attention: one wave per (b, n, head); f=16 queries/keys; bias tab.
// Row layout stays the x-layout; the (b f) n <-> (b n) f permutation is index math.
// ---------------------------------------------------------------------------
__global__ __launch_bounds__(32) void attnt_kernel(const bf16_t* __restrict__ q,
                                                   const bf16_t* __restrict__ k,
                                                   const bf16_t* __restrict__ v,
                                                   const float* __restrict__ tab,
                                                   bf16_t* __restrict__ o) {
    const int s = blockIdx.x;           // 0..511  (b, n)
    const int h = blockIdx.y;           // 0..19
    const int b = s >> 8, n = s & 255;
    const int lane = threadIdx.x, fl = lane & 15, kb = (lane >> 4) * 8, rofs = 8 * (lane >> 4);
    __shared__ bf16_t Ps[16][40];       // [query][key(32, padded)]
    __shared__ bf16_t Vs[64][40];       // [dh][key(32, padded)]

    v16bf qf[2], kf[2];
#pragma unroll
    for (int kc = 0; kc < 2; ++kc) {
        const bf16_t* rp = ((size_t)((b * F_VID + fl) * N_TOK + n)) * D_MODEL + h * DH + kc * 32 + kb + q;
        qf[kc] = frag16_scaled(rp, QK_SCALE);
        const bf16_t* rk = ((size_t)((b * F_VID + fl) * N_TOK + n)) * D_MODEL + h * DH + kc * 32 + kb + k;
        kf[kc] = frag16(rk);
    }
    v8f S = {};
    S = wmma_bf16(qf[0], kf[0], S);
    S = wmma_bf16(qf[1], kf[1], S);
    // bias + softmax over 16 keys
#pragma unroll
    for (int r = 0; r < 8; ++r) {
        float sv = S[r] + tab[(rofs + r - fl + F_VID - 1) * HEADS + h];
        float rm = redmax16(sv);
        float pv = expf(sv - rm);
        float rs = redsum16(pv);
        pv /= rs;
        Ps[rofs + r][fl]      = (bf16_t)pv;
        Ps[rofs + r][16 + fl] = (bf16_t)0.f;
    }
    // stage V transposed, zero-pad keys 16..31
    {
        int fj = lane >> 1, c0 = (lane & 1) * 32;
        const bf16_t* vp = v + ((size_t)((b * F_VID + fj) * N_TOK + n)) * D_MODEL + h * DH + c0;
#pragma unroll
        for (int e = 0; e < 32; ++e) Vs[c0 + e][fj] = vp[e];
    }
    for (int d = lane; d < 64; d += 32)
#pragma unroll
        for (int kk = 16; kk < 32; ++kk) Vs[d][kk] = (bf16_t)0.f;
    __syncthreads();

    v16bf pf = frag16(&Ps[fl][kb]);
#pragma unroll
    for (int nj = 0; nj < 4; ++nj) {
        v16bf vfr = frag16(&Vs[nj * 16 + fl][kb]);
        v8f a2 = {};
        a2 = wmma_bf16(pf, vfr, a2);
#pragma unroll
        for (int r = 0; r < 8; ++r) {
            size_t grow = (size_t)((b * F_VID + rofs + r) * N_TOK + n);
            o[grow * D_MODEL + (size_t)h * DH + nj * 16 + fl] = (bf16_t)a2[r];
        }
    }
}

// ---------------------------------------------------------------------------
// Relative-position-bias MLP: tab[31][20]
// ---------------------------------------------------------------------------
__global__ __launch_bounds__(640) void posbias_kernel(const float* __restrict__ pb1w,
                                                      const float* __restrict__ pb1b,
                                                      const float* __restrict__ pb2w,
                                                      const float* __restrict__ pb2b,
                                                      const float* __restrict__ pb3w,
                                                      const float* __restrict__ pb3b,
                                                      float* __restrict__ tab) {
    __shared__ float h1[640];
    __shared__ float h2[640];
    const int row = blockIdx.x;                 // 0..30
    const float rel = (float)(row - (F_VID - 1));
    const int t = threadIdx.x;
    float a = rel * pb1w[t] + pb1b[t];
    h1[t] = a / (1.f + expf(-a));
    __syncthreads();
    float sum = pb2b[t];
    for (int i = 0; i < 640; ++i) sum += h1[i] * pb2w[i * 640 + t];
    h2[t] = sum / (1.f + expf(-sum));
    __syncthreads();
    if (t < HEADS) {
        float s3 = pb3b[t];
        for (int i = 0; i < 640; ++i) s3 += h2[i] * pb3w[i * HEADS + t];
        tab[row * HEADS + t] = s3;
    }
}

// ---------------------------------------------------------------------------
// host side
// ---------------------------------------------------------------------------
extern "C" void kernel_launch(void* const* d_in, const int* in_sizes, int n_in,
                              void* d_out, int out_size, void* d_ws, size_t ws_size,
                              hipStream_t stream) {
    const float* hs    = (const float*)d_in[0];
    const float* enc   = (const float*)d_in[1];
    const float* n1w   = (const float*)d_in[2];
    const float* n1b   = (const float*)d_in[3];
    const float* a1q   = (const float*)d_in[4];
    const float* a1k   = (const float*)d_in[5];
    const float* a1v   = (const float*)d_in[6];
    const float* a1ow  = (const float*)d_in[7];
    const float* a1ob  = (const float*)d_in[8];
    const float* n2w   = (const float*)d_in[9];
    const float* n2b   = (const float*)d_in[10];
    const float* a2q   = (const float*)d_in[11];
    const float* a2k   = (const float*)d_in[12];
    const float* a2v   = (const float*)d_in[13];
    const float* a2ow  = (const float*)d_in[14];
    const float* a2ob  = (const float*)d_in[15];
    const float* n3w   = (const float*)d_in[16];
    const float* n3b   = (const float*)d_in[17];
    const float* ff1w  = (const float*)d_in[18];
    const float* ff1b  = (const float*)d_in[19];
    const float* ff2w  = (const float*)d_in[20];
    const float* ff2b  = (const float*)d_in[21];
    const float* ntw   = (const float*)d_in[22];
    const float* ntb   = (const float*)d_in[23];
    const float* atq   = (const float*)d_in[24];
    const float* atk   = (const float*)d_in[25];
    const float* atv   = (const float*)d_in[26];
    const float* atow  = (const float*)d_in[27];
    const float* atob  = (const float*)d_in[28];
    const float* pb1w  = (const float*)d_in[29];
    const float* pb1b  = (const float*)d_in[30];
    const float* pb2w  = (const float*)d_in[31];
    const float* pb2b  = (const float*)d_in[32];
    const float* pb3w  = (const float*)d_in[33];
    const float* pb3b  = (const float*)d_in[34];
    (void)in_sizes; (void)n_in; (void)out_size; (void)ws_size;

    const int M = M_TOT, D = D_MODEL, DC = DC_ENC, ME = ME_TOT;

    uintptr_t base = (uintptr_t)d_ws;
    auto take = [&](size_t bytes) -> void* {
        void* p = (void*)base;
        base += (bytes + 255) & ~(size_t)255;
        return p;
    };
    float*  x    = (float*) take((size_t)M * D * 4);
    bf16_t* nx   = (bf16_t*)take((size_t)M * D * 2);
    bf16_t* qb   = (bf16_t*)take((size_t)M * D * 2);
    bf16_t* kbf  = (bf16_t*)take((size_t)M * D * 2);
    bf16_t* vbf  = (bf16_t*)take((size_t)M * D * 2);
    bf16_t* ob   = (bf16_t*)take((size_t)M * D * 2);
    bf16_t* encb = (bf16_t*)take((size_t)ME * DC * 2);
    bf16_t* kkb  = (bf16_t*)take((size_t)ME * D * 2);
    bf16_t* vvb  = (bf16_t*)take((size_t)ME * D * 2);
    bf16_t* ffm  = (bf16_t*)take((size_t)M * FF_HALF * 2);
    float*  tab  = (float*) take(31 * HEADS * 4);
    bf16_t* wA1q = (bf16_t*)take((size_t)D * D * 2);
    bf16_t* wA1k = (bf16_t*)take((size_t)D * D * 2);
    bf16_t* wA1v = (bf16_t*)take((size_t)D * D * 2);
    bf16_t* wA1o = (bf16_t*)take((size_t)D * D * 2);
    bf16_t* wA2q = (bf16_t*)take((size_t)D * D * 2);
    bf16_t* wA2k = (bf16_t*)take((size_t)DC * D * 2);
    bf16_t* wA2v = (bf16_t*)take((size_t)DC * D * 2);
    bf16_t* wA2o = (bf16_t*)take((size_t)D * D * 2);
    bf16_t* wFF1 = (bf16_t*)take((size_t)D * 2 * FF_HALF * 2);
    bf16_t* wFF2 = (bf16_t*)take((size_t)FF_HALF * D * 2);
    bf16_t* wAtq = (bf16_t*)take((size_t)D * D * 2);
    bf16_t* wAtk = (bf16_t*)take((size_t)D * D * 2);
    bf16_t* wAtv = (bf16_t*)take((size_t)D * D * 2);
    bf16_t* wAto = (bf16_t*)take((size_t)D * D * 2);

    auto cvtT = [&](const float* W, bf16_t* WT, int K, int N) {
        int n = K * N;
        cvtT_kernel<<<(n + 255) / 256, 256, 0, stream>>>(W, WT, K, N);
    };
    // weight conversions (fp32 -> transposed bf16)
    cvtT(a1q, wA1q, D, D);   cvtT(a1k, wA1k, D, D);
    cvtT(a1v, wA1v, D, D);   cvtT(a1ow, wA1o, D, D);
    cvtT(a2q, wA2q, D, D);   cvtT(a2k, wA2k, DC, D);
    cvtT(a2v, wA2v, DC, D);  cvtT(a2ow, wA2o, D, D);
    cvtT(ff1w, wFF1, D, 2 * FF_HALF);
    cvtT(ff2w, wFF2, FF_HALF, D);
    cvtT(atq, wAtq, D, D);   cvtT(atk, wAtk, D, D);
    cvtT(atv, wAtv, D, D);   cvtT(atow, wAto, D, D);
    cvt_kernel<<<(ME * DC + 255) / 256, 256, 0, stream>>>(enc, encb, ME * DC);
    hipMemcpyAsync(x, hs, (size_t)M * D * 4, hipMemcpyDeviceToDevice, stream);

    const dim3 gemmG(D / 128, M / 128);          // 10 x 64
    const dim3 gemmGE(D / 128, (ME + 127) / 128);// 10 x 20

    // ---- attn1 (sparse-causal self-attention) ----
    ln_kernel<<<M, 256, 0, stream>>>(x, n1w, n1b, nx, D);
    gemm_kernel<<<gemmG, 128, 0, stream>>>(nx, wA1q, nullptr, nullptr, nullptr, qb, M, D, D);
    gemm_kernel<<<gemmG, 128, 0, stream>>>(nx, wA1k, nullptr, nullptr, nullptr, kbf, M, D, D);
    gemm_kernel<<<gemmG, 128, 0, stream>>>(nx, wA1v, nullptr, nullptr, nullptr, vbf, M, D, D);
    attn_kernel<<<dim3(BF_TOT, HEADS), 128, 0, stream>>>(qb, kbf, vbf, ob, 0, 2 * N_TOK);
    gemm_kernel<<<gemmG, 128, 0, stream>>>(ob, wA1o, a1ob, x, x, nullptr, M, D, D);

    // ---- attn2 (cross-attention) ----
    ln_kernel<<<M, 256, 0, stream>>>(x, n2w, n2b, nx, D);
    gemm_kernel<<<gemmG, 128, 0, stream>>>(nx, wA2q, nullptr, nullptr, nullptr, qb, M, D, D);
    gemm_kernel<<<gemmGE, 128, 0, stream>>>(encb, wA2k, nullptr, nullptr, nullptr, kkb, ME, D, DC);
    gemm_kernel<<<gemmGE, 128, 0, stream>>>(encb, wA2v, nullptr, nullptr, nullptr, vvb, ME, D, DC);
    attn_kernel<<<dim3(BF_TOT, HEADS), 128, 0, stream>>>(qb, kkb, vvb, ob, 1, NK_CROSS);
    gemm_kernel<<<gemmG, 128, 0, stream>>>(ob, wA2o, a2ob, x, x, nullptr, M, D, D);

    // ---- feed-forward (GEGLU) ----
    ln_kernel<<<M, 256, 0, stream>>>(x, n3w, n3b, nx, D);
    geglu_kernel<<<dim3(FF_HALF / 128, M / 128), 128, 0, stream>>>(nx, wFF1, ff1b, ffm, M, FF_HALF, D);
    gemm_kernel<<<gemmG, 128, 0, stream>>>(ffm, wFF2, ff2b, x, x, nullptr, M, D, FF_HALF);

    // ---- temporal attention ----
    posbias_kernel<<<31, 640, 0, stream>>>(pb1w, pb1b, pb2w, pb2b, pb3w, pb3b, tab);
    ln_kernel<<<M, 256, 0, stream>>>(x, ntw, ntb, nx, D);
    gemm_kernel<<<gemmG, 128, 0, stream>>>(nx, wAtq, nullptr, nullptr, nullptr, qb, M, D, D);
    gemm_kernel<<<gemmG, 128, 0, stream>>>(nx, wAtk, nullptr, nullptr, nullptr, kbf, M, D, D);
    gemm_kernel<<<gemmG, 128, 0, stream>>>(nx, wAtv, nullptr, nullptr, nullptr, vbf, M, D, D);
    attnt_kernel<<<dim3(2 * N_TOK, HEADS), 32, 0, stream>>>(qb, kbf, vbf, tab, ob);
    gemm_kernel<<<gemmG, 128, 0, stream>>>(ob, wAto, atob, x, (float*)d_out, nullptr, M, D, D);
}
